// SeqDecoder_15341623182075
// MI455X (gfx1250) — matrix operand order repaired and checked by reading, not compiled, not run on hardware
//
#include <hip/hip_runtime.h>
#include <hip/hip_bf16.h>
#include <stdint.h>

typedef uint32_t u32;
typedef __attribute__((ext_vector_type(4)))  u32    v4u;
typedef __attribute__((ext_vector_type(8)))  u32    v8u;
typedef __attribute__((ext_vector_type(8)))  float  v8f;
typedef __attribute__((ext_vector_type(16))) __bf16 v16bf;

constexpr int BATCH = 256;
constexpr int HID   = 1024;
constexpr int EMB   = 256;
constexpr int POSE  = 108;
constexpr int STEPS = 128;
constexpr int KT    = HID / 32;            // 32 k-tiles of K=32
constexpr int MT    = BATCH / 16;          // 16 m-tiles
constexpr int HALF_DW = MT * KT * 32 * 8;  // 131072 dwords: one (hi or lo) packed h plane
constexpr int SLOT_DW = 2 * HALF_DW;       // 262144 dwords per hs slot (hi+lo)

// ---------- bf16 split helpers (manual RNE, no __bf16 arithmetic needed) ----------
__device__ __forceinline__ u32 f2bf(float x) {
    u32 u = __builtin_bit_cast(u32, x);
    return (u + 0x7FFFu + ((u >> 16) & 1u)) >> 16;
}
__device__ __forceinline__ float bf2f(u32 b) { return __builtin_bit_cast(float, b << 16); }
// pack two floats into one dword of bf16 pairs, producing hi and lo planes
__device__ __forceinline__ void split2(float a, float b, u32& hiD, u32& loD) {
    u32 ah = f2bf(a), bh = f2bf(b);
    u32 al = f2bf(a - bf2f(ah)), bl = f2bf(b - bf2f(bh));
    hiD = ah | (bh << 16);
    loD = al | (bl << 16);
}

__device__ __forceinline__ v8f wmma_bf(v8u a, v8u b, v8f c) {
    return __builtin_amdgcn_wmma_f32_16x16x32_bf16(
        false, __builtin_bit_cast(v16bf, a),
        false, __builtin_bit_cast(v16bf, b),
        (short)0, c, false, false);
}

__device__ __forceinline__ float sigm(float x) { return 1.0f / (1.0f + __expf(-x)); }

// ---------- prep: pack W_hh [4096,1024] into WMMA B-operand layout, hi/lo planes ----------
// B layout (16x16x32 bf16): lane L<16 -> col N = nt*16+L, K = kt*32 + {0..7, 16..23} in dword pairs;
//                           lane L>=16 -> col N = nt*16+L-16, K = kt*32 + {8..15, 24..31}.
__global__ __launch_bounds__(256) void k_pack_whh(const float* __restrict__ W,
                                                  u32* __restrict__ Hi, u32* __restrict__ Lo) {
    int idx  = blockIdx.x * 256 + threadIdx.x;       // < 32*256*32
    int lane = idx & 31;
    int nt   = (idx >> 5) & 255;
    int kt   = idx >> 13;
    int n    = nt * 16 + (lane & 15);
    int kb   = kt * 32 + ((lane >> 4) << 3);
    const float* src = W + (size_t)n * HID + kb;
    u32 hd[8], ld[8];
#pragma unroll
    for (int v = 0; v < 8; ++v) split2(src[2 * v + (v >> 2) * 8], src[2 * v + (v >> 2) * 8 + 1], hd[v], ld[v]);
    size_t off = ((size_t)(kt * 256 + nt) * 32 + lane) * 8;
    v4u h0v = {hd[0], hd[1], hd[2], hd[3]}, h1v = {hd[4], hd[5], hd[6], hd[7]};
    v4u l0v = {ld[0], ld[1], ld[2], ld[3]}, l1v = {ld[4], ld[5], ld[6], ld[7]};
    *(v4u*)(Hi + off) = h0v; *(v4u*)(Hi + off + 4) = h1v;
    *(v4u*)(Lo + off) = l0v; *(v4u*)(Lo + off + 4) = l1v;
}

// ---------- prep: pack W_fc [108,1024] (padded to 112 cols) ----------
__global__ __launch_bounds__(256) void k_pack_wfc(const float* __restrict__ W,
                                                  u32* __restrict__ Hi, u32* __restrict__ Lo) {
    int idx = blockIdx.x * 256 + threadIdx.x;
    if (idx >= 32 * 7 * 32) return;
    int lane = idx & 31;
    int nt   = (idx >> 5) % 7;
    int kt   = idx / (7 * 32);
    int n    = nt * 16 + (lane & 15);
    int kb   = kt * 32 + ((lane >> 4) << 3);
    u32 hd[8], ld[8];
#pragma unroll
    for (int v = 0; v < 8; ++v) {
        float a = (n < POSE) ? W[(size_t)n * HID + kb + 2 * v + (v >> 2) * 8] : 0.0f;
        float b = (n < POSE) ? W[(size_t)n * HID + kb + 2 * v + (v >> 2) * 8 + 1] : 0.0f;
        split2(a, b, hd[v], ld[v]);
    }
    size_t off = ((size_t)(kt * 7 + nt) * 32 + lane) * 8;
    v4u h0v = {hd[0], hd[1], hd[2], hd[3]}, h1v = {hd[4], hd[5], hd[6], hd[7]};
    v4u l0v = {ld[0], ld[1], ld[2], ld[3]}, l1v = {ld[4], ld[5], ld[6], ld[7]};
    *(v4u*)(Hi + off) = h0v; *(v4u*)(Hi + off + 4) = h1v;
    *(v4u*)(Lo + off) = l0v; *(v4u*)(Lo + off + 4) = l1v;
}

// ---------- prep: x_part = dec_input @ W_ih^T + b_ih + b_hh  [256,4096] fp32 ----------
__global__ __launch_bounds__(256) void k_xpart(const float* __restrict__ dec,
                                               const float* __restrict__ Wih,
                                               const float* __restrict__ bih,
                                               const float* __restrict__ bhh,
                                               float* __restrict__ xp) {
    int idx = blockIdx.x * 256 + threadIdx.x;   // < 256*4096
    int b = idx >> 12, n = idx & 4095;
    const float4* dv = (const float4*)(dec + (size_t)b * EMB);
    const float4* wv = (const float4*)(Wih + (size_t)n * EMB);
    float s = bih[n] + bhh[n];
#pragma unroll 4
    for (int e = 0; e < EMB / 4; ++e) {
        float4 d = dv[e], w = wv[e];
        s += d.x * w.x + d.y * w.y + d.z * w.z + d.w * w.w;
    }
    xp[idx] = s;
}

// ---------- prep: pack h0 into hs slot 0 (A-operand layout, hi/lo) ----------
// A layout (16x16x32 bf16): lane L<16 -> row M=mt*16+L, K = kt*32 + {0..7, 16..23};
//                           lane L>=16 -> row M=mt*16+L-16, K = kt*32 + {8..15, 24..31}.
__global__ __launch_bounds__(256) void k_pack_h0(const float* __restrict__ h0,
                                                 u32* __restrict__ hsP) {
    int idx  = blockIdx.x * 256 + threadIdx.x;   // < 16*32*32
    int mt   = idx >> 10;
    int kt   = (idx >> 5) & 31;
    int lane = idx & 31;
    int m    = mt * 16 + (lane & 15);
    int ka   = kt * 32 + ((lane >> 4) << 3);
    const float* s1 = h0 + (size_t)m * HID + ka;
    const float* s2 = s1 + 16;
    u32 hd[8], ld[8];
#pragma unroll
    for (int i = 0; i < 4; ++i) split2(s1[2 * i], s1[2 * i + 1], hd[i], ld[i]);
#pragma unroll
    for (int i = 0; i < 4; ++i) split2(s2[2 * i], s2[2 * i + 1], hd[4 + i], ld[4 + i]);
    size_t fo = ((size_t)(mt * KT + kt) * 32 + lane) * 8;
    v4u h0v = {hd[0], hd[1], hd[2], hd[3]}, h1v = {hd[4], hd[5], hd[6], hd[7]};
    v4u l0v = {ld[0], ld[1], ld[2], ld[3]}, l1v = {ld[4], ld[5], ld[6], ld[7]};
    *(v4u*)(hsP + fo) = h0v;           *(v4u*)(hsP + fo + 4) = h1v;
    *(v4u*)(hsP + HALF_DW + fo) = l0v; *(v4u*)(hsP + HALF_DW + fo + 4) = l1v;
}

__global__ __launch_bounds__(256) void k_copy(const float* __restrict__ src,
                                              float* __restrict__ dst) {
    int idx = blockIdx.x * 256 + threadIdx.x;
    dst[idx] = src[idx];
}

// ---------- LSTM step: gates = x_part + h @ W_hh^T ; pointwise ; pack h_new ----------
// grid = 128 blocks: blockIdx = jt (0..63, 16 hidden cols) x mh (0..1, 128 batch rows)
// 8 waves/block, wave owns one m-tile x all 4 gates (pointwise stays wave-local).
__global__ __launch_bounds__(256) void k_step(u32* __restrict__ hsP,
                                              const u32* __restrict__ WpHi,
                                              const u32* __restrict__ WpLo,
                                              const float* __restrict__ xp,
                                              float* __restrict__ cbuf,
                                              int t,
                                              float* __restrict__ hf,
                                              float* __restrict__ cf) {
    __shared__ float lds[8 * 16 * 16];
    int jt   = blockIdx.x & 63;
    int mh   = blockIdx.x >> 6;
    int wave = threadIdx.x >> 5;
    int lane = threadIdx.x & 31;
    int mtg  = mh * 8 + wave;

    const u32* A = hsP + (size_t)t * SLOT_DW;
    v8f acc[4] = {};
    for (int kt = 0; kt < KT; ++kt) {
        size_t fo = ((size_t)(mtg * KT + kt) * 32 + lane) * 8;
        v8u ahi = *(const v8u*)(A + fo);
        v8u alo = *(const v8u*)(A + HALF_DW + fo);
#pragma unroll
        for (int g = 0; g < 4; ++g) {
            size_t bo = ((size_t)(kt * 256 + g * 64 + jt) * 32 + lane) * 8;
            v8u bhi = *(const v8u*)(WpHi + bo);
            v8u blo = *(const v8u*)(WpLo + bo);
            acc[g] = wmma_bf(ahi, bhi, acc[g]);
            acc[g] = wmma_bf(ahi, blo, acc[g]);
            acc[g] = wmma_bf(alo, bhi, acc[g]);
        }
    }

    // pointwise: C layout -> VGPR r: row = r (lanes<16) / r+8 (lanes>=16); col = lane%16
    int colL  = lane & 15;
    int col   = jt * 16 + colL;
    int rbase = (lane >> 4) * 8;
    float* ldsW = lds + wave * 256;
#pragma unroll
    for (int r = 0; r < 8; ++r) {
        int row = mtg * 16 + rbase + r;
        const float* xr = xp + (size_t)row * 4096 + col;
        float gi = acc[0][r] + xr[0];
        float gf = acc[1][r] + xr[1024];
        float gg = acc[2][r] + xr[2048];
        float go = acc[3][r] + xr[3072];
        float I = sigm(gi), F = sigm(gf), G = tanhf(gg), O = sigm(go);
        float cold = cbuf[(size_t)row * HID + col];
        float cn = F * cold + I * G;
        float hn = O * tanhf(cn);
        cbuf[(size_t)row * HID + col] = cn;
        ldsW[(rbase + r) * 16 + colL] = hn;
        if (t == STEPS - 1) {
            hf[(size_t)row * HID + col] = hn;
            cf[(size_t)row * HID + col] = cn;
        }
    }
    __syncthreads();

    // transpose through LDS and write h_new pre-packed (A layout) into slot t+1
    {
        int rowL = lane & 15;
        int c0   = (lane >> 4) * 8;                 // K-offset group within the 16-col block
        const float* s = ldsW + rowL * 16 + c0;
        u32 hd[4], ld4[4];
#pragma unroll
        for (int i = 0; i < 4; ++i) split2(s[2 * i], s[2 * i + 1], hd[i], ld4[i]);
        int kt2 = jt >> 1;
        int vb  = (jt & 1) * 4;                      // K 0..15 -> dwords 0..3 ; K 16..31 -> 4..7
        size_t fo = ((size_t)(mtg * KT + kt2) * 32 + lane) * 8 + vb;
        u32* dst = hsP + (size_t)(t + 1) * SLOT_DW;
        v4u hv = {hd[0], hd[1], hd[2], hd[3]};
        v4u lv = {ld4[0], ld4[1], ld4[2], ld4[3]};
        *(v4u*)(dst + fo) = hv;
        *(v4u*)(dst + HALF_DW + fo) = lv;
    }
}

// ---------- output projection: outputs = hs @ W_fc^T + b_fc ----------
__global__ __launch_bounds__(256) void k_proj(const u32* __restrict__ hsP,
                                              const u32* __restrict__ FHi,
                                              const u32* __restrict__ FLo,
                                              const float* __restrict__ bfc,
                                              float* __restrict__ out) {
    int wave = threadIdx.x >> 5, lane = threadIdx.x & 31;
    int mtid = blockIdx.x * 8 + wave;               // 0..2047 global m-tiles (rows = T*B)
    int slot = (mtid >> 4) + 1;
    int mt   = mtid & 15;
    const u32* A = hsP + (size_t)slot * SLOT_DW;
    v8f acc[7] = {};
    for (int kt = 0; kt < KT; ++kt) {
        size_t fo = ((size_t)(mt * KT + kt) * 32 + lane) * 8;
        v8u ahi = *(const v8u*)(A + fo);
        v8u alo = *(const v8u*)(A + HALF_DW + fo);
#pragma unroll
        for (int nt = 0; nt < 7; ++nt) {
            size_t bo = ((size_t)(kt * 7 + nt) * 32 + lane) * 8;
            v8u bhi = *(const v8u*)(FHi + bo);
            v8u blo = *(const v8u*)(FLo + bo);
            acc[nt] = wmma_bf(ahi, bhi, acc[nt]);
            acc[nt] = wmma_bf(ahi, blo, acc[nt]);
            acc[nt] = wmma_bf(alo, bhi, acc[nt]);
        }
    }
    int rbase = (lane >> 4) * 8, cl = lane & 15;
#pragma unroll
    for (int nt = 0; nt < 7; ++nt) {
        int p = nt * 16 + cl;
        if (p < POSE) {
            float bb = bfc[p];
#pragma unroll
            for (int r = 0; r < 8; ++r) {
                int row = mtid * 16 + rbase + r;
                out[(size_t)row * POSE + p] = acc[nt][r] + bb;
            }
        }
    }
}

extern "C" void kernel_launch(void* const* d_in, const int* in_sizes, int n_in,
                              void* d_out, int out_size, void* d_ws, size_t ws_size,
                              hipStream_t stream) {
    (void)in_sizes; (void)n_in; (void)out_size; (void)ws_size;
    const float* h0  = (const float*)d_in[0];
    const float* c0  = (const float*)d_in[1];
    const float* dec = (const float*)d_in[2];
    const float* Wih = (const float*)d_in[3];
    const float* Whh = (const float*)d_in[4];
    const float* bih = (const float*)d_in[5];
    const float* bhh = (const float*)d_in[6];
    const float* Wfc = (const float*)d_in[7];
    const float* bfc = (const float*)d_in[8];
    // d_in[9] = num_steps (always 128 for this problem)

    char* ws = (char*)d_ws;
    size_t o = 0;
    u32* hsP  = (u32*)(ws + o); o += (size_t)(STEPS + 1) * SLOT_DW * 4;   // 135.3 MB
    u32* WpHi = (u32*)(ws + o); o += (size_t)KT * 256 * 32 * 8 * 4;       // 8 MB
    u32* WpLo = (u32*)(ws + o); o += (size_t)KT * 256 * 32 * 8 * 4;       // 8 MB
    u32* FHi  = (u32*)(ws + o); o += (size_t)KT * 7 * 32 * 8 * 4;         // 224 KB
    u32* FLo  = (u32*)(ws + o); o += (size_t)KT * 7 * 32 * 8 * 4;         // 224 KB
    float* xp   = (float*)(ws + o); o += (size_t)BATCH * 4 * HID * 4;     // 4 MB
    float* cbuf = (float*)(ws + o); o += (size_t)BATCH * HID * 4;         // 1 MB

    float* out = (float*)d_out;
    float* hf  = out + (size_t)STEPS * BATCH * POSE;
    float* cf  = hf + (size_t)BATCH * HID;

    k_pack_whh<<<1024, 256, 0, stream>>>(Whh, WpHi, WpLo);
    k_pack_wfc<<<28, 256, 0, stream>>>(Wfc, FHi, FLo);
    k_xpart<<<4096, 256, 0, stream>>>(dec, Wih, bih, bhh, xp);
    k_pack_h0<<<64, 256, 0, stream>>>(h0, hsP);
    k_copy<<<1024, 256, 0, stream>>>(c0, cbuf);
    for (int t = 0; t < STEPS; ++t)
        k_step<<<128, 256, 0, stream>>>(hsP, WpHi, WpLo, xp, cbuf, t, hf, cf);
    k_proj<<<256, 256, 0, stream>>>(hsP, FHi, FLo, bfc, out);
}
